// Net_53858889891907
// MI455X (gfx1250) — matrix operand order repaired and checked by reading, not compiled, not run on hardware
//
#include <hip/hip_runtime.h>
#include <math.h>

typedef __attribute__((ext_vector_type(16))) __bf16 v16bf;
typedef __attribute__((ext_vector_type(8)))  float  v8f;

#define NEG_SLOPE 0.2f

// ---------- helpers ----------
__device__ __forceinline__ __bf16 f2bf(float f) {
  union { float f; unsigned u; } in; in.f = f;
  unsigned u = (in.u + 0x7FFFu + ((in.u >> 16) & 1u)) >> 16;  // RNE
  union { unsigned short s; __bf16 b; } out; out.s = (unsigned short)u;
  return out.b;
}

// float atomic max via signed/unsigned ordering trick (valid with -inf init)
__device__ __forceinline__ void atomicMaxF(float* addr, float v) {
  if (v >= 0.0f) atomicMax((int*)addr, __float_as_int(v));
  else           atomicMin((unsigned int*)addr, __float_as_uint(v));
}

__device__ __forceinline__ void edge_sd(const long long* __restrict__ ei,
                                        long long E, long long e, int& s, int& d) {
  if (e < E) { s = (int)ei[e]; d = (int)ei[E + e]; }
  else       { s = d = (int)(e - E); }   // appended self-loops
}

// ---------- generic fill ----------
__global__ void k_fill(float* __restrict__ p, float v, long long n) {
  long long i = (long long)blockIdx.x * blockDim.x + threadIdx.x;
  if (i < n) p[i] = v;
}

// ---------- pack W1 [512,64] fp32 -> bf16 B-fragments ----------
// fragment index frag = kt*4+ct ; per-lane 16 contiguous bf16:
// elem j -> K = kt*32 + (j<8 ? (lane/16)*8 + j : 16 + (lane/16)*8 + j-8) , col = ct*16 + lane%16
__global__ void k_pack_w1(const float* __restrict__ W1, __bf16* __restrict__ wp) {
  int t = blockIdx.x * blockDim.x + threadIdx.x;
  if (t >= 16 * 4 * 32 * 16) return;
  int j    = t & 15;
  int lane = (t >> 4) & 31;
  int frag = t >> 9;
  int ct = frag & 3, kt = frag >> 2;
  int lm = lane & 15, half = lane >> 4;
  int col = ct * 16 + lm;
  int K   = kt * 32 + (j < 8 ? half * 8 + j : 16 + half * 8 + (j - 8));
  wp[t] = f2bf(W1[(size_t)K * 64 + col]);
}

// ---------- h1 = x @ W1 via WMMA bf16 (fp32 accum) ----------
// one wave per 16-row tile; computes the full 16x64 output block:
// A is loaded+converted once per K step and reused by 4 WMMAs (one per
// 16-column tile), K=512 -> 16 steps x 4 WMMA = 64 WMMA per wave.
__global__ void k_gemm1(const float* __restrict__ x, const __bf16* __restrict__ wp,
                        float* __restrict__ h1, int nRowTiles) {
  int wave = blockIdx.x * (blockDim.x >> 5) + (threadIdx.x >> 5);
  int lane = threadIdx.x & 31;
  if (wave >= nRowTiles) return;
  int lm = lane & 15, half = lane >> 4;
  const float* arow = x + (size_t)(wave * 16 + lm) * 512;
  v8f acc0 = {}, acc1 = {}, acc2 = {}, acc3 = {};
#pragma unroll
  for (int kt = 0; kt < 16; ++kt) {
    int kb = kt * 32 + half * 8;
    if (kt < 15) __builtin_prefetch(arow + kb + 32, 0, 1);  // global_prefetch next K step
    const float4 a0 = *(const float4*)(arow + kb);
    const float4 a1 = *(const float4*)(arow + kb + 4);
    const float4 a2 = *(const float4*)(arow + kb + 16);
    const float4 a3 = *(const float4*)(arow + kb + 20);
    v16bf a;
    a[0]  = f2bf(a0.x); a[1]  = f2bf(a0.y); a[2]  = f2bf(a0.z); a[3]  = f2bf(a0.w);
    a[4]  = f2bf(a1.x); a[5]  = f2bf(a1.y); a[6]  = f2bf(a1.z); a[7]  = f2bf(a1.w);
    a[8]  = f2bf(a2.x); a[9]  = f2bf(a2.y); a[10] = f2bf(a2.z); a[11] = f2bf(a2.w);
    a[12] = f2bf(a3.x); a[13] = f2bf(a3.y); a[14] = f2bf(a3.z); a[15] = f2bf(a3.w);
    const __bf16* bp = wp + (size_t)kt * 4 * 512 + lane * 16;
    v16bf b0 = *(const v16bf*)(bp);
    v16bf b1 = *(const v16bf*)(bp + 512);
    v16bf b2 = *(const v16bf*)(bp + 1024);
    v16bf b3 = *(const v16bf*)(bp + 1536);
    acc0 = __builtin_amdgcn_wmma_f32_16x16x32_bf16(false, a, false, b0, (short)0, acc0, false, false);
    acc1 = __builtin_amdgcn_wmma_f32_16x16x32_bf16(false, a, false, b1, (short)0, acc1, false, false);
    acc2 = __builtin_amdgcn_wmma_f32_16x16x32_bf16(false, a, false, b2, (short)0, acc2, false, false);
    acc3 = __builtin_amdgcn_wmma_f32_16x16x32_bf16(false, a, false, b3, (short)0, acc3, false, false);
  }
#pragma unroll
  for (int r = 0; r < 8; ++r) {
    float* orow = h1 + (size_t)(wave * 16 + half * 8 + r) * 64 + lm;
    orow[0]  = acc0[r];
    orow[16] = acc1[r];
    orow[32] = acc2[r];
    orow[48] = acc3[r];
  }
}

// ---------- per-node attention logits, layer 1 ----------
__global__ void k_alpha1(const float* __restrict__ h1, const float* __restrict__ as,
                         const float* __restrict__ ad, float* __restrict__ asrc,
                         float* __restrict__ adst, int N) {
  int t = blockIdx.x * blockDim.x + threadIdx.x;
  if (t >= N * 8) return;
  int n = t >> 3, h = t & 7;
  const float* row = h1 + (size_t)n * 64 + h * 8;
  float s = 0.f, d = 0.f;
#pragma unroll
  for (int i = 0; i < 8; ++i) { s += row[i] * as[h * 8 + i]; d += row[i] * ad[h * 8 + i]; }
  asrc[t] = s; adst[t] = d;
}

// ---------- edge passes, layer 1 (per edge x head) ----------
__global__ void k_edge_max1(const long long* __restrict__ ei, long long E, long long Etot,
                            const float* __restrict__ asrc, const float* __restrict__ adst,
                            float* __restrict__ emax) {
  long long t = (long long)blockIdx.x * blockDim.x + threadIdx.x;
  if (t >= Etot * 8) return;
  long long e = t >> 3; int h = (int)(t & 7);
  int s, d; edge_sd(ei, E, e, s, d);
  float v = asrc[(size_t)s * 8 + h] + adst[(size_t)d * 8 + h];
  v = v > 0.f ? v : NEG_SLOPE * v;
  atomicMaxF(&emax[(size_t)d * 8 + h], v);
}

__global__ void k_edge_sum1(const long long* __restrict__ ei, long long E, long long Etot,
                            const float* __restrict__ asrc, const float* __restrict__ adst,
                            const float* __restrict__ emax, float* __restrict__ denom) {
  long long t = (long long)blockIdx.x * blockDim.x + threadIdx.x;
  if (t >= Etot * 8) return;
  long long e = t >> 3; int h = (int)(t & 7);
  int s, d; edge_sd(ei, E, e, s, d);
  float v = asrc[(size_t)s * 8 + h] + adst[(size_t)d * 8 + h];
  v = v > 0.f ? v : NEG_SLOPE * v;
  atomicAdd(&denom[(size_t)d * 8 + h], __expf(v - emax[(size_t)d * 8 + h]));
}

__global__ void k_edge_scat1(const long long* __restrict__ ei, long long E, long long Etot,
                             const float* __restrict__ asrc, const float* __restrict__ adst,
                             const float* __restrict__ emax, const float* __restrict__ denom,
                             const float* __restrict__ h1, float* __restrict__ out1) {
  long long t = (long long)blockIdx.x * blockDim.x + threadIdx.x;
  if (t >= Etot * 8) return;
  long long e = t >> 3; int h = (int)(t & 7);
  int s, d; edge_sd(ei, E, e, s, d);
  float v = asrc[(size_t)s * 8 + h] + adst[(size_t)d * 8 + h];
  v = v > 0.f ? v : NEG_SLOPE * v;
  float alpha = __expf(v - emax[(size_t)d * 8 + h]) / denom[(size_t)d * 8 + h];
  const float* hs = h1 + (size_t)s * 64 + h * 8;
  float* od = out1 + (size_t)d * 64 + h * 8;
  const float4 p0 = *(const float4*)(hs);
  const float4 p1 = *(const float4*)(hs + 4);
  atomicAdd(&od[0], p0.x * alpha); atomicAdd(&od[1], p0.y * alpha);
  atomicAdd(&od[2], p0.z * alpha); atomicAdd(&od[3], p0.w * alpha);
  atomicAdd(&od[4], p1.x * alpha); atomicAdd(&od[5], p1.y * alpha);
  atomicAdd(&od[6], p1.z * alpha); atomicAdd(&od[7], p1.w * alpha);
}

// ---------- bias + ELU (in place) ----------
__global__ void k_bias_elu(float* __restrict__ p, const float* __restrict__ b, long long n) {
  long long t = (long long)blockIdx.x * blockDim.x + threadIdx.x;
  if (t >= n) return;
  float v = p[t] + b[t & 63];
  p[t] = v > 0.f ? v : (__expf(v) - 1.0f);
}

// ---------- layer-2 linear (64->10) + attention logits ----------
__global__ void k_lin2(const float* __restrict__ h, const float* __restrict__ W2,
                       const float* __restrict__ as2, const float* __restrict__ ad2,
                       float* __restrict__ h2, float* __restrict__ asrc2,
                       float* __restrict__ adst2, int N) {
  int n = blockIdx.x * blockDim.x + threadIdx.x;
  if (n >= N) return;
  float acc[10];
#pragma unroll
  for (int c = 0; c < 10; ++c) acc[c] = 0.f;
  const float* row = h + (size_t)n * 64;
  for (int j = 0; j < 64; ++j) {
    float v = row[j];
    const float* w = W2 + j * 10;
#pragma unroll
    for (int c = 0; c < 10; ++c) acc[c] += v * w[c];
  }
  float s = 0.f, d = 0.f;
#pragma unroll
  for (int c = 0; c < 10; ++c) {
    h2[(size_t)n * 10 + c] = acc[c];
    s += acc[c] * as2[c];
    d += acc[c] * ad2[c];
  }
  asrc2[n] = s; adst2[n] = d;
}

// ---------- edge passes, layer 2 (single head, per edge) ----------
__global__ void k_edge_max2(const long long* __restrict__ ei, long long E, long long Etot,
                            const float* __restrict__ asrc, const float* __restrict__ adst,
                            float* __restrict__ emax) {
  long long e = (long long)blockIdx.x * blockDim.x + threadIdx.x;
  if (e >= Etot) return;
  int s, d; edge_sd(ei, E, e, s, d);
  float v = asrc[s] + adst[d];
  v = v > 0.f ? v : NEG_SLOPE * v;
  atomicMaxF(&emax[d], v);
}

__global__ void k_edge_sum2(const long long* __restrict__ ei, long long E, long long Etot,
                            const float* __restrict__ asrc, const float* __restrict__ adst,
                            const float* __restrict__ emax, float* __restrict__ denom) {
  long long e = (long long)blockIdx.x * blockDim.x + threadIdx.x;
  if (e >= Etot) return;
  int s, d; edge_sd(ei, E, e, s, d);
  float v = asrc[s] + adst[d];
  v = v > 0.f ? v : NEG_SLOPE * v;
  atomicAdd(&denom[d], __expf(v - emax[d]));
}

__global__ void k_edge_scat2(const long long* __restrict__ ei, long long E, long long Etot,
                             const float* __restrict__ asrc, const float* __restrict__ adst,
                             const float* __restrict__ emax, const float* __restrict__ denom,
                             const float* __restrict__ h2, float* __restrict__ out) {
  long long e = (long long)blockIdx.x * blockDim.x + threadIdx.x;
  if (e >= Etot) return;
  int s, d; edge_sd(ei, E, e, s, d);
  float v = asrc[s] + adst[d];
  v = v > 0.f ? v : NEG_SLOPE * v;
  float alpha = __expf(v - emax[d]) / denom[d];
  const float* hs = h2 + (size_t)s * 10;
  float* od = out + (size_t)d * 10;
#pragma unroll
  for (int c = 0; c < 10; ++c) atomicAdd(&od[c], hs[c] * alpha);
}

// ---------- bias + log_softmax over 10 classes (in place on d_out) ----------
__global__ void k_logsoftmax(float* __restrict__ out, const float* __restrict__ b2, int N) {
  int n = blockIdx.x * blockDim.x + threadIdx.x;
  if (n >= N) return;
  float v[10]; float m = -INFINITY;
#pragma unroll
  for (int c = 0; c < 10; ++c) { v[c] = out[(size_t)n * 10 + c] + b2[c]; m = fmaxf(m, v[c]); }
  float s = 0.f;
#pragma unroll
  for (int c = 0; c < 10; ++c) s += __expf(v[c] - m);
  float ls = __logf(s);
#pragma unroll
  for (int c = 0; c < 10; ++c) out[(size_t)n * 10 + c] = v[c] - m - ls;
}

// ---------- driver ----------
extern "C" void kernel_launch(void* const* d_in, const int* in_sizes, int n_in,
                              void* d_out, int out_size, void* d_ws, size_t ws_size,
                              hipStream_t stream) {
  const float*     x   = (const float*)d_in[0];
  const long long* ei  = (const long long*)d_in[1];
  const float*     W1  = (const float*)d_in[2];
  const float*     as1 = (const float*)d_in[3];
  const float*     ad1 = (const float*)d_in[4];
  const float*     b1  = (const float*)d_in[5];
  const float*     W2  = (const float*)d_in[6];
  const float*     as2 = (const float*)d_in[7];
  const float*     ad2 = (const float*)d_in[8];
  const float*     b2  = (const float*)d_in[9];
  float* out = (float*)d_out;

  const int F = 512;
  int       N    = in_sizes[0] / F;
  long long E    = (long long)in_sizes[1] / 2;
  long long Etot = E + N;

  char* ws = (char*)d_ws;
  size_t off = 0;
  auto alloc = [&](size_t bytes) -> void* {
    void* p = ws + off;
    off = (off + bytes + 255) & ~(size_t)255;
    return p;
  };
  __bf16* wp    = (__bf16*)alloc((size_t)16 * 4 * 32 * 16 * 2);
  float* h1     = (float*)alloc((size_t)N * 64 * 4);
  float* asrc1  = (float*)alloc((size_t)N * 8 * 4);
  float* adst1  = (float*)alloc((size_t)N * 8 * 4);
  float* emax1  = (float*)alloc((size_t)N * 8 * 4);
  float* denom1 = (float*)alloc((size_t)N * 8 * 4);
  float* out1   = (float*)alloc((size_t)N * 64 * 4);
  float* h2     = (float*)alloc((size_t)N * 10 * 4);
  float* asrc2  = (float*)alloc((size_t)N * 4);
  float* adst2  = (float*)alloc((size_t)N * 4);
  float* emax2  = (float*)alloc((size_t)N * 4);
  float* denom2 = (float*)alloc((size_t)N * 4);
  (void)ws_size; (void)n_in; (void)out_size;

  const int B = 256;
  auto nb = [&](long long n) { return (unsigned)((n + B - 1) / B); };

  // layer 1
  k_pack_w1<<<nb(16 * 4 * 32 * 16), B, 0, stream>>>(W1, wp);
  int nRowTiles = (N + 15) / 16;
  k_gemm1<<<(unsigned)((nRowTiles + 7) / 8), 256, 0, stream>>>(x, wp, h1, nRowTiles);
  k_alpha1<<<nb((long long)N * 8), B, 0, stream>>>(h1, as1, ad1, asrc1, adst1, N);

  k_fill<<<nb((long long)N * 8), B, 0, stream>>>(emax1, -INFINITY, (long long)N * 8);
  k_fill<<<nb((long long)N * 8), B, 0, stream>>>(denom1, 0.f, (long long)N * 8);
  k_fill<<<nb((long long)N * 64), B, 0, stream>>>(out1, 0.f, (long long)N * 64);

  k_edge_max1 <<<nb(Etot * 8), B, 0, stream>>>(ei, E, Etot, asrc1, adst1, emax1);
  k_edge_sum1 <<<nb(Etot * 8), B, 0, stream>>>(ei, E, Etot, asrc1, adst1, emax1, denom1);
  k_edge_scat1<<<nb(Etot * 8), B, 0, stream>>>(ei, E, Etot, asrc1, adst1, emax1, denom1, h1, out1);

  k_bias_elu<<<nb((long long)N * 64), B, 0, stream>>>(out1, b1, (long long)N * 64);

  // layer 2
  k_lin2<<<nb(N), B, 0, stream>>>(out1, W2, as2, ad2, h2, asrc2, adst2, N);

  k_fill<<<nb(N), B, 0, stream>>>(emax2, -INFINITY, N);
  k_fill<<<nb(N), B, 0, stream>>>(denom2, 0.f, N);
  k_fill<<<nb((long long)N * 10), B, 0, stream>>>(out, 0.f, (long long)N * 10);

  k_edge_max2 <<<nb(Etot), B, 0, stream>>>(ei, E, Etot, asrc2, adst2, emax2);
  k_edge_sum2 <<<nb(Etot), B, 0, stream>>>(ei, E, Etot, asrc2, adst2, emax2, denom2);
  k_edge_scat2<<<nb(Etot), B, 0, stream>>>(ei, E, Etot, asrc2, adst2, emax2, denom2, h2, out);

  k_logsoftmax<<<nb(N), B, 0, stream>>>(out, b2, N);
}
